// MultiHeadPooledSelfAttention_13615046328884
// MI455X (gfx1250) — matrix operand order repaired and checked by reading, not compiled, not run on hardware
//
#include <hip/hip_runtime.h>

typedef unsigned short u16;
typedef __attribute__((ext_vector_type(16))) __bf16 v16bf;
typedef __attribute__((ext_vector_type(8)))  float  v8f;

// ---------- small helpers ----------
__device__ inline u16 f2bf(float f) {
    union { float f; unsigned u; } v; v.f = f;
    unsigned u = v.u;
    unsigned r = (u + 0x7FFFu + ((u >> 16) & 1u)) >> 16;   // RNE
    return (u16)r;
}
__device__ inline float bf2f(u16 b) {
    union { unsigned u; float f; } v; v.u = ((unsigned)b) << 16;
    return v.f;
}

union BF8 { uint4 u; __bf16 h[8]; };

__device__ inline v16bf mk_frag(uint4 lo4, uint4 hi4) {
    BF8 lo, hi; lo.u = lo4; hi.u = hi4;
    v16bf r;
#pragma unroll
    for (int i = 0; i < 8; ++i) { r[i] = lo.h[i]; r[8 + i] = hi.h[i]; }
    return r;
}

// A-matrix fragment (16x32 bf16): lane m = lane&15, half h = lane>>4.
// lane holds K = 16h..16h+15 contiguous.
__device__ inline v16bf loadA(const u16* base, int ld, int m0, int k0, int lane) {
    int m = m0 + (lane & 15), h = lane >> 4;
    const u16* p = base + (size_t)m * ld + k0 + 16 * h;
    return mk_frag(*(const uint4*)p, *(const uint4*)(p + 8));
}

// B-matrix fragment (32x16 bf16) from row-major (N,K) source:
// lane n = lane&15, half h = lane>>4; lane holds K = 16h..16h+15 contiguous.
__device__ inline v16bf loadB(const u16* base, int ld, int n0, int k0, int lane) {
    int n = n0 + (lane & 15), h = lane >> 4;
    const u16* p = base + (size_t)n * ld + k0 + 16 * h;
    const uint4* q = (const uint4*)p;
    return mk_frag(q[0], q[1]);
}

__device__ inline v8f wmma_bf16(v16bf a, v16bf b, v8f c) {
    return __builtin_amdgcn_wmma_f32_16x16x32_bf16(false, a, false, b,
                                                   (short)0, c, false, false);
}
__device__ inline void zero8(v8f& a) {
#pragma unroll
    for (int i = 0; i < 8; ++i) a[i] = 0.0f;
}

// ---------- constants ----------
#define BATCH 4
#define LSEQ  1025      // 1 + 32*32
#define LP    1040      // padded rows per head (multiple of 16)
#define NHEAD 8
#define HDIM  512
#define QKV   4096
#define IDIM  512
#define MROWS 4100      // B * LSEQ

// ---------- f32 -> bf16 convert ----------
__global__ void cvt_bf16_kernel(const float* __restrict__ src,
                                u16* __restrict__ dst, int n) {
    int i = blockIdx.x * 256 + threadIdx.x;
    if (i < n) dst[i] = f2bf(src[i]);
}

// ---------- QKV projection GEMM: Y = X @ W^T + b ----------
// A: Xb (MROWS_pad x 512), W: (4096 x 512) row-major. Output bf16 in
// (B, head, L, 512) layout. Per wave: 32x64 tile (2x4 fragments).
__global__ __launch_bounds__(256)
void gemm_qkv_kernel(const u16* __restrict__ A, const u16* __restrict__ W,
                     const float* __restrict__ bias, u16* __restrict__ Out,
                     int M) {
    int tid = threadIdx.x, lane = tid & 31, wid = tid >> 5;
    int m_base = blockIdx.x * 64 + (wid >> 2) * 32;
    int n_base = blockIdx.y * 256 + (wid & 3) * 64;
    v8f acc[2][4];
#pragma unroll
    for (int i = 0; i < 2; ++i)
#pragma unroll
        for (int j = 0; j < 4; ++j) zero8(acc[i][j]);
    for (int k0 = 0; k0 < IDIM; k0 += 32) {
        if (k0 + 32 < IDIM)
            __builtin_prefetch(A + (size_t)(m_base + (lane & 15)) * IDIM + k0 + 32, 0, 1);
        v16bf a0 = loadA(A, IDIM, m_base,      k0, lane);
        v16bf a1 = loadA(A, IDIM, m_base + 16, k0, lane);
        v16bf b0 = loadB(W, IDIM, n_base,      k0, lane);
        v16bf b1 = loadB(W, IDIM, n_base + 16, k0, lane);
        v16bf b2 = loadB(W, IDIM, n_base + 32, k0, lane);
        v16bf b3 = loadB(W, IDIM, n_base + 48, k0, lane);
        acc[0][0] = wmma_bf16(a0, b0, acc[0][0]);
        acc[0][1] = wmma_bf16(a0, b1, acc[0][1]);
        acc[0][2] = wmma_bf16(a0, b2, acc[0][2]);
        acc[0][3] = wmma_bf16(a0, b3, acc[0][3]);
        acc[1][0] = wmma_bf16(a1, b0, acc[1][0]);
        acc[1][1] = wmma_bf16(a1, b1, acc[1][1]);
        acc[1][2] = wmma_bf16(a1, b2, acc[1][2]);
        acc[1][3] = wmma_bf16(a1, b3, acc[1][3]);
    }
    int h = lane >> 4, nn = lane & 15;
#pragma unroll
    for (int i = 0; i < 2; ++i)
#pragma unroll
        for (int j = 0; j < 4; ++j) {
            int n = n_base + j * 16 + nn;
            float bv = bias[n];
            int head = n >> 9, c = n & 511;
#pragma unroll
            for (int e = 0; e < 8; ++e) {
                int m = m_base + i * 16 + e + 8 * h;
                if (m < M) {
                    int b = m / LSEQ, l = m - b * LSEQ;
                    Out[(((size_t)(b * NHEAD + head)) * LSEQ + l) * HDIM + c] =
                        f2bf(acc[i][j][e] + bv);
                }
            }
        }
}

// ---------- Pool GEMM over spatial tokens ----------
// mode 0: plain (PK). mode 1: + 2D sincos pos-embed (PQ).
// mode 2: transposed output PV^T [(bh*512 + c)*LP + l] (PV).
// modes 0/1 write [(bh*LP + l)*512 + c]. Per wave: 32x64 tile.
__global__ __launch_bounds__(256)
void gemm_pool_kernel(const u16* __restrict__ T, const u16* __restrict__ Wp,
                      const float* __restrict__ bias, u16* __restrict__ Out,
                      int mode) {
    int tid = threadIdx.x, lane = tid & 31, wid = tid >> 5;
    int r_base = blockIdx.x * 64;          // 64-row tile, never crosses a head
    int bh = r_base >> 10;
    int l0 = 1 + (r_base & 1023);
    const u16* Abase = T + ((size_t)bh * LSEQ + l0) * HDIM;
    int m_loc  = (wid >> 2) * 32;
    int n_base = blockIdx.y * 256 + (wid & 3) * 64;
    v8f acc[2][4];
#pragma unroll
    for (int i = 0; i < 2; ++i)
#pragma unroll
        for (int j = 0; j < 4; ++j) zero8(acc[i][j]);
    for (int k0 = 0; k0 < HDIM; k0 += 32) {
        v16bf a0 = loadA(Abase, HDIM, m_loc,      k0, lane);
        v16bf a1 = loadA(Abase, HDIM, m_loc + 16, k0, lane);
        v16bf b0 = loadB(Wp, HDIM, n_base,      k0, lane);
        v16bf b1 = loadB(Wp, HDIM, n_base + 16, k0, lane);
        v16bf b2 = loadB(Wp, HDIM, n_base + 32, k0, lane);
        v16bf b3 = loadB(Wp, HDIM, n_base + 48, k0, lane);
        acc[0][0] = wmma_bf16(a0, b0, acc[0][0]);
        acc[0][1] = wmma_bf16(a0, b1, acc[0][1]);
        acc[0][2] = wmma_bf16(a0, b2, acc[0][2]);
        acc[0][3] = wmma_bf16(a0, b3, acc[0][3]);
        acc[1][0] = wmma_bf16(a1, b0, acc[1][0]);
        acc[1][1] = wmma_bf16(a1, b1, acc[1][1]);
        acc[1][2] = wmma_bf16(a1, b2, acc[1][2]);
        acc[1][3] = wmma_bf16(a1, b3, acc[1][3]);
    }
    int h = lane >> 4, nn = lane & 15;
#pragma unroll
    for (int i = 0; i < 2; ++i)
#pragma unroll
        for (int j = 0; j < 4; ++j) {
            int c = n_base + j * 16 + nn;
            float bv = bias[c];
#pragma unroll
            for (int e = 0; e < 8; ++e) {
                int l = l0 + m_loc + i * 16 + e + 8 * h;
                float val = acc[i][j][e] + bv;
                if (mode == 1) {
                    // emb[p, c]: p = l-1; sectors of 128: sin(y), cos(y), sin(x), cos(x)
                    int p = l - 1;
                    float pos = (c < 256) ? (float)(p >> 5) : (float)(p & 31);
                    int jj = c & 127;
                    float omega = __expf(-(float)jj * 0.07195578415606394f); // ln(1e4)/128
                    float ang = pos * omega;
                    val += ((c >> 7) & 1) ? __cosf(ang) : __sinf(ang);
                }
                if (mode == 2)
                    Out[((size_t)bh * HDIM + c) * LP + l] = f2bf(val);
                else
                    Out[((size_t)bh * LP + l) * HDIM + c] = f2bf(val);
            }
        }
}

// ---------- cls token passthrough ----------
__global__ void cls_copy_kernel(const u16* __restrict__ Qb, const u16* __restrict__ Kb,
                                const u16* __restrict__ Vb, u16* __restrict__ PQ,
                                u16* __restrict__ PK, u16* __restrict__ PVt) {
    int t = blockIdx.x * 256 + threadIdx.x;
    if (t >= 32 * HDIM) return;
    int bh = t >> 9, c = t & 511;
    size_t src = (size_t)bh * LSEQ * HDIM + c;
    PQ[(size_t)bh * LP * HDIM + c]  = Qb[src];
    PK[(size_t)bh * LP * HDIM + c]  = Kb[src];
    PVt[((size_t)bh * HDIM + c) * LP + 0] = Vb[src];
}

// ---------- fused attention: logits -> softmax -> @PV^T + residual ----------
// grid (65 qtiles, 8 heads, 4 batch), 256 threads (8 waves), dyn LDS 116224 B
__global__ __launch_bounds__(256)
void attention_kernel(const u16* __restrict__ PQ, const u16* __restrict__ PK,
                      const u16* __restrict__ PVt, const u16* __restrict__ Qb,
                      u16* __restrict__ SA) {
    extern __shared__ char smem[];
    u16*   sQ    = (u16*)smem;                                // 16x512 bf16 = 16384 B
    float* sLog  = (float*)(smem + 16384);                    // 16xLP f32  = 66560 B
    u16*   sAttn = (u16*)(smem + 16384 + 16 * LP * 4);        // 16xLP bf16 = 33280 B

    int tid = threadIdx.x, lane = tid & 31, wid = tid >> 5;
    int bh = blockIdx.z * NHEAD + blockIdx.y;
    int q0 = blockIdx.x * 16;
    const u16* pq  = PQ  + (size_t)bh * LP * HDIM;
    const u16* pk  = PK  + (size_t)bh * LP * HDIM;
    const u16* pvt = PVt + (size_t)bh * HDIM * LP;

    // stage PQ q-tile into LDS (16 rows x 512 bf16 = 1024 uint4)
    {
        const uint4* src = (const uint4*)(pq + (size_t)q0 * HDIM);
        uint4* dst = (uint4*)sQ;
        for (int i = tid; i < 1024; i += 256) dst[i] = src[i];
    }
    __syncthreads();

    const float scale = 0.04419417382415922f;   // 1/sqrt(512)
    const int ktiles = (LSEQ + 15) / 16;        // 65

    // ---- logits: S = PQ @ PK^T ----
    // Each wave owns adjacent key-tile pairs (t0, t0+1) so the LDS A-fragment
    // is reused across two WMMAs. t0 = 2*wid + 16*p covers evens 0..64; the
    // phantom tile 65 reads stay inside the workspace and its store is guarded.
    for (int t0 = wid * 2; t0 < 66; t0 += 16) {
        int k0a = t0 * 16;
        int k0b = (t0 + 1) * 16;
        v8f acc0, acc1; zero8(acc0); zero8(acc1);
        for (int kk = 0; kk < HDIM; kk += 32) {
            v16bf a  = loadA(sQ, HDIM, 0, kk, lane);       // ds_read_b128 x2
            v16bf b0 = loadB(pk, HDIM, k0a, kk, lane);
            v16bf b1 = loadB(pk, HDIM, k0b, kk, lane);
            acc0 = wmma_bf16(a, b0, acc0);
            acc1 = wmma_bf16(a, b1, acc1);
        }
        int n = lane & 15, h = lane >> 4;
#pragma unroll
        for (int e = 0; e < 8; ++e) {
            int m = e + 8 * h;
            sLog[m * LP + k0a + n] = acc0[e] * scale;
            if (t0 + 1 < ktiles) sLog[m * LP + k0b + n] = acc1[e] * scale;
        }
    }
    __syncthreads();

    // ---- softmax: 2 rows per wave, cross-lane reduce with shfl_xor ----
    for (int rr = 0; rr < 2; ++rr) {
        int r = 2 * wid + rr;
        float mx = -1e30f;
        for (int k = lane; k < LSEQ; k += 32) mx = fmaxf(mx, sLog[r * LP + k]);
        for (int off = 16; off; off >>= 1) mx = fmaxf(mx, __shfl_xor(mx, off, 32));
        float sum = 0.0f;
        for (int k = lane; k < LSEQ; k += 32) {
            float e = __expf(sLog[r * LP + k] - mx);
            sLog[r * LP + k] = e;
            sum += e;
        }
        for (int off = 16; off; off >>= 1) sum += __shfl_xor(sum, off, 32);
        float inv = 1.0f / sum;
        for (int k = lane; k < LP; k += 32)
            sAttn[r * LP + k] = (k < LSEQ) ? f2bf(sLog[r * LP + k] * inv) : (u16)0;
    }
    __syncthreads();

    // ---- O = attn @ PV : wave owns 64 output dims (4 n-tiles) ----
    v8f acc[4];
    zero8(acc[0]); zero8(acc[1]); zero8(acc[2]); zero8(acc[3]);
    for (int t = 0; t < ktiles; ++t) {
        int k0 = t * 16;
        v16bf a = loadA(sAttn, LP, 0, k0, lane);          // attn tile from LDS
#pragma unroll
        for (int j = 0; j < 4; ++j) {
            int c0 = wid * 64 + j * 16;
            v16bf b = loadB(pvt, LP, c0, k0, lane);       // PV^T rows contiguous
            acc[j] = wmma_bf16(a, b, acc[j]);
        }
    }

    // ---- epilogue: residual (+Q on spatial tokens) and stacked store ----
    int b = bh >> 3, head = bh & 7;
    int nn = lane & 15, h = lane >> 4;
#pragma unroll
    for (int j = 0; j < 4; ++j) {
        int c = wid * 64 + j * 16 + nn;
#pragma unroll
        for (int e = 0; e < 8; ++e) {
            int l = q0 + e + 8 * h;
            if (l < LSEQ) {
                float v = acc[j][e];
                if (l >= 1) v += bf2f(Qb[((size_t)bh * LSEQ + l) * HDIM + c]);
                SA[((size_t)(b * LSEQ + l)) * QKV + head * HDIM + c] = f2bf(v);
            }
        }
    }
}

// ---------- output projection: out = SA @ Wd^T + bd (f32 out) ----------
__global__ __launch_bounds__(256)
void gemm_final_kernel(const u16* __restrict__ A, const u16* __restrict__ W,
                       const float* __restrict__ bias, float* __restrict__ Out,
                       int M) {
    int tid = threadIdx.x, lane = tid & 31, wid = tid >> 5;
    int m_base = blockIdx.x * 64 + (wid >> 2) * 32;
    int n_base = blockIdx.y * 128 + (wid & 3) * 32;
    v8f acc[2][2];
    zero8(acc[0][0]); zero8(acc[0][1]); zero8(acc[1][0]); zero8(acc[1][1]);
    for (int k0 = 0; k0 < QKV; k0 += 32) {
        v16bf a0 = loadA(A, QKV, m_base,      k0, lane);
        v16bf a1 = loadA(A, QKV, m_base + 16, k0, lane);
        v16bf b0 = loadB(W, QKV, n_base,      k0, lane);
        v16bf b1 = loadB(W, QKV, n_base + 16, k0, lane);
        acc[0][0] = wmma_bf16(a0, b0, acc[0][0]);
        acc[0][1] = wmma_bf16(a0, b1, acc[0][1]);
        acc[1][0] = wmma_bf16(a1, b0, acc[1][0]);
        acc[1][1] = wmma_bf16(a1, b1, acc[1][1]);
    }
    int h = lane >> 4, nn = lane & 15;
#pragma unroll
    for (int i = 0; i < 2; ++i)
#pragma unroll
        for (int j = 0; j < 2; ++j) {
            int n = n_base + j * 16 + nn;
            float bv = bias[n];
#pragma unroll
            for (int e = 0; e < 8; ++e) {
                int m = m_base + i * 16 + e + 8 * h;
                if (m < M) Out[(size_t)m * 512 + n] = acc[i][j][e] + bv;
            }
        }
}

// ==================== host launch ====================
extern "C" void kernel_launch(void* const* d_in, const int* in_sizes, int n_in,
                              void* d_out, int out_size, void* d_ws, size_t ws_size,
                              hipStream_t stream) {
    const float* x   = (const float*)d_in[0];
    const float* Wq  = (const float*)d_in[1];
    const float* bq  = (const float*)d_in[2];
    const float* Wk  = (const float*)d_in[3];
    const float* bk  = (const float*)d_in[4];
    const float* Wv  = (const float*)d_in[5];
    const float* bv  = (const float*)d_in[6];
    const float* Wpq = (const float*)d_in[7];
    const float* bpq = (const float*)d_in[8];
    const float* Wpk = (const float*)d_in[9];
    const float* bpk = (const float*)d_in[10];
    const float* Wpv = (const float*)d_in[11];
    const float* bpv = (const float*)d_in[12];
    const float* Wd  = (const float*)d_in[13];
    const float* bd  = (const float*)d_in[14];
    float* out = (float*)d_out;

    char* ws = (char*)d_ws;
    size_t off = 0;
    auto alloc = [&](size_t bytes) {
        size_t o = off;
        off = (off + bytes + 255) & ~(size_t)255;
        return o;
    };
    const int MPAD = 65 * 64;  // 4160 rows (GEMM grid coverage)
    size_t oXB   = alloc((size_t)MPAD * IDIM * 2);
    size_t oWQ   = alloc((size_t)QKV * IDIM * 2);
    size_t oWK   = alloc((size_t)QKV * IDIM * 2);
    size_t oWV   = alloc((size_t)QKV * IDIM * 2);
    size_t oWPQ  = alloc((size_t)HDIM * HDIM * 2);
    size_t oWPK  = alloc((size_t)HDIM * HDIM * 2);
    size_t oWPV  = alloc((size_t)HDIM * HDIM * 2);
    size_t oWD   = alloc((size_t)IDIM * QKV * 2);
    size_t oQ    = alloc((size_t)32 * LSEQ * HDIM * 2);
    size_t oK    = alloc((size_t)32 * LSEQ * HDIM * 2);
    size_t oV    = alloc((size_t)32 * LSEQ * HDIM * 2);
    size_t oPQ   = alloc((size_t)32 * LP * HDIM * 2);
    size_t oPK   = alloc((size_t)32 * LP * HDIM * 2);
    size_t oPVT  = alloc((size_t)32 * HDIM * LP * 2);
    size_t oSA   = alloc((size_t)MPAD * QKV * 2);

    u16* Xb   = (u16*)(ws + oXB);
    u16* Wqb  = (u16*)(ws + oWQ);
    u16* Wkb  = (u16*)(ws + oWK);
    u16* Wvb  = (u16*)(ws + oWV);
    u16* Wpqb = (u16*)(ws + oWPQ);
    u16* Wpkb = (u16*)(ws + oWPK);
    u16* Wpvb = (u16*)(ws + oWPV);
    u16* Wdb  = (u16*)(ws + oWD);
    u16* Qb   = (u16*)(ws + oQ);
    u16* Kb   = (u16*)(ws + oK);
    u16* Vb   = (u16*)(ws + oV);
    u16* PQb  = (u16*)(ws + oPQ);
    u16* PKb  = (u16*)(ws + oPK);
    u16* PVtb = (u16*)(ws + oPVT);
    u16* SAb  = (u16*)(ws + oSA);

    auto cvt = [&](const float* s, u16* d, int n) {
        cvt_bf16_kernel<<<(n + 255) / 256, 256, 0, stream>>>(s, d, n);
    };
    cvt(x,   Xb,   MROWS * IDIM);
    cvt(Wq,  Wqb,  QKV * IDIM);
    cvt(Wk,  Wkb,  QKV * IDIM);
    cvt(Wv,  Wvb,  QKV * IDIM);
    cvt(Wpq, Wpqb, HDIM * HDIM);
    cvt(Wpk, Wpkb, HDIM * HDIM);
    cvt(Wpv, Wpvb, HDIM * HDIM);
    cvt(Wd,  Wdb,  IDIM * QKV);

    // QKV projections: block tile 64 x 256 (8 waves of 32x64)
    dim3 gq(65, QKV / 256);
    gemm_qkv_kernel<<<gq, 256, 0, stream>>>(Xb, Wqb, bq, Qb, MROWS);
    gemm_qkv_kernel<<<gq, 256, 0, stream>>>(Xb, Wkb, bk, Kb, MROWS);
    gemm_qkv_kernel<<<gq, 256, 0, stream>>>(Xb, Wvb, bv, Vb, MROWS);

    // Pooling (spatial tokens); PQ fuses pos-embed; PV stored transposed
    dim3 gp(512, HDIM / 256);
    gemm_pool_kernel<<<gp, 256, 0, stream>>>(Qb, Wpqb, bpq, PQb, 1);
    gemm_pool_kernel<<<gp, 256, 0, stream>>>(Kb, Wpkb, bpk, PKb, 0);
    gemm_pool_kernel<<<gp, 256, 0, stream>>>(Vb, Wpvb, bpv, PVtb, 2);

    // cls passthrough
    cls_copy_kernel<<<64, 256, 0, stream>>>(Qb, Kb, Vb, PQb, PKb, PVtb);

    // fused attention
    dim3 ga(65, NHEAD, BATCH);
    size_t lds = 16384 + (size_t)16 * LP * 4 + (size_t)16 * LP * 2;  // 116224 B
    attention_kernel<<<ga, 256, lds, stream>>>(PQb, PKb, PVtb, Qb, SAb);

    // output projection (small: keep 260 blocks for occupancy)
    dim3 gf(65, 4);
    gemm_final_kernel<<<gf, 256, 0, stream>>>(SAb, Wdb, bd, out, MROWS);
}
